// AdapterFusionBlock_87256555585591
// MI455X (gfx1250) — matrix operand order repaired and checked by reading, not compiled, not run on hardware
//
#include <hip/hip_runtime.h>
#include <hip/hip_bf16.h>
#include <math.h>

typedef __attribute__((ext_vector_type(16))) _Float16 v16h;
typedef __attribute__((ext_vector_type(8)))  _Float16 v8h;
typedef __attribute__((ext_vector_type(8)))  float    v8f;

// ---------------- problem constants ----------------
#define DIMC   768
#define NHC    12
#define HDC    64
#define WSC    16
#define NTOK   32768      // B*H*W tokens
#define NWIN   128        // 8 * 4 * 4 windows
#define BHCNT  1536       // NWIN * NHC
#define MLPD_C 3072
#define AD_C   576

// ---------------- WMMA helpers ----------------
__device__ __forceinline__ v8f wmma16x16x32(v16h a, v16h b, v8f c) {
  return __builtin_amdgcn_wmma_f32_16x16x32_f16(false, a, false, b, (short)0, c,
                                                false, false);
}

__device__ __forceinline__ v16h make_frag(v8h lo, v8h hi) {
  v16h r;
#pragma unroll
  for (int i = 0; i < 8; ++i) { r[i] = lo[i]; r[i + 8] = hi[i]; }
  return r;
}

// A fragment (16x32 f16): lane L -> row = L%16; halves hold K = kb..kb+7 and
// kb+16..kb+23 where kb = (L/16)*8  (ISA 7.12.2 16-bit A layout)
__device__ __forceinline__ v16h load_frag_a(const _Float16* base, int lda,
                                            int m0, int k0, int lane) {
  int row = m0 + (lane & 15);
  int kb  = k0 + ((lane >> 4) << 3);
  const _Float16* p = base + (size_t)row * lda + kb;
  v8h lo = *(const v8h*)p;
  v8h hi = *(const v8h*)(p + 16);
  return make_frag(lo, hi);
}

// B fragment (32x16 f16) from transposed weight Wt[N][K]: lane L -> col = L%16,
// halves hold K = ks..ks+7 and ks+16..ks+23, ks = (L/16)*8 (matches A layout)
__device__ __forceinline__ v16h load_frag_b(const _Float16* wt, int ldb,
                                            int n0, int k0, int lane) {
  int col = n0 + (lane & 15);
  int ks  = k0 + ((lane >> 4) << 3);
  const _Float16* p = wt + (size_t)col * ldb + ks;
  v8h lo = *(const v8h*)p;
  v8h hi = *(const v8h*)(p + 16);
  return make_frag(lo, hi);
}

// ---------------- async global -> LDS (CDNA5, ASYNCcnt-tracked) ----------
__device__ __forceinline__ void async_g2l_b128(unsigned lds_addr,
                                               const void* gaddr) {
  unsigned long long ga = (unsigned long long)(size_t)gaddr;
  asm volatile("global_load_async_to_lds_b128 %0, %1, off"
               :: "v"(lds_addr), "v"(ga) : "memory");
}
__device__ __forceinline__ void wait_async_le2() {
  asm volatile("s_wait_asynccnt 0x2" ::: "memory");
}
__device__ __forceinline__ void wait_async_le1() {
  asm volatile("s_wait_asynccnt 0x1" ::: "memory");
}
__device__ __forceinline__ void wait_async_le0() {
  asm volatile("s_wait_asynccnt 0x0" ::: "memory");
}

// ---------------- misc device helpers ----------------
__device__ __forceinline__ float gelu_erf(float v) {
  return 0.5f * v * (1.0f + erff(v * 0.70710678118654752f));
}
__device__ __forceinline__ float sigmoidf_(float z) {
  return 1.0f / (1.0f + __expf(-z));
}

// global token index -> window-token index
__device__ __forceinline__ int g2m(int g) {
  int b  = g >> 12;
  int hh = (g >> 6) & 63;
  int ww = g & 63;
  int wh = hh >> 4, qh = hh & 15;
  int wi = ww >> 4, qw = ww & 15;
  int win = (b << 4) + (wh << 2) + wi;
  return (win << 8) + (qh << 4) + qw;
}
// window-token index -> global token index
__device__ __forceinline__ int m2g(int m) {
  int win = m >> 8, pos = m & 255;
  int qh = pos >> 4, qw = pos & 15;
  int b  = win >> 4;
  int wh = (win >> 2) & 3, wi = win & 3;
  int hh = (wh << 4) + qh, ww = (wi << 4) + qw;
  return (b << 12) + (hh << 6) + ww;
}

// ---------------- weight convert + transpose (f32 [K][N] -> f16 [N][K]) ------
__global__ void cvt_wt_k(const float* __restrict__ src, _Float16* __restrict__ dst,
                         int K, int N) {
  long id = (long)blockIdx.x * 256 + threadIdx.x;
  long tot = (long)K * N;
  if (id >= tot) return;
  int k = (int)(id / N), n = (int)(id % N);
  dst[(size_t)n * K + k] = (_Float16)src[id];
}

// ---------------- layernorm (768) -> f16, optional window-order scatter ------
__global__ __launch_bounds__(256) void ln_k(const float* __restrict__ x,
                                            const float* __restrict__ w,
                                            const float* __restrict__ b,
                                            _Float16* __restrict__ out,
                                            int winorder) {
  int g = blockIdx.x;
  int t = threadIdx.x;
  const float* xr = x + (size_t)g * DIMC;
  float v0 = xr[t], v1 = xr[t + 256], v2 = xr[t + 512];
  __shared__ float rs_[256], rq_[256];
  rs_[t] = v0 + v1 + v2;
  rq_[t] = v0 * v0 + v1 * v1 + v2 * v2;
  __syncthreads();
  for (int off = 128; off > 0; off >>= 1) {
    if (t < off) { rs_[t] += rs_[t + off]; rq_[t] += rq_[t + off]; }
    __syncthreads();
  }
  float mean = rs_[0] * (1.0f / 768.0f);
  float var  = rq_[0] * (1.0f / 768.0f) - mean * mean;
  float rst  = rsqrtf(var + 1e-5f);
  int m = winorder ? g2m(g) : g;
  _Float16* orow = out + (size_t)m * DIMC;
  orow[t]       = (_Float16)((v0 - mean) * rst * w[t]       + b[t]);
  orow[t + 256] = (_Float16)((v1 - mean) * rst * w[t + 256] + b[t + 256]);
  orow[t + 512] = (_Float16)((v2 - mean) * rst * w[t + 512] + b[t + 512]);
}

// ---------------- shuffle-adapter reductions ----------------
// stats layout (floats): [0,3071] m0[b][c]; [3072..3079] mu[b]; [3080..3087] var[b]
__global__ __launch_bounds__(384) void adapter_stats_k(const _Float16* __restrict__ xn,
                                                       float* __restrict__ stats) {
  int b = blockIdx.x;
  int c = threadIdx.x;   // 0..383
  float s0 = 0.f, s1 = 0.f, s1q = 0.f;
  for (int pos = 0; pos < 4096; ++pos) {
    int g = (b << 12) + pos;
    int m = g2m(g);
    float x0 = (float)xn[(size_t)m * DIMC + c];
    float x1 = (float)xn[(size_t)m * DIMC + 384 + c];
    s0 += x0; s1 += x1; s1q += x1 * x1;
  }
  stats[b * 384 + c] = s0 * (1.0f / 4096.0f);
  __shared__ float r1[384], r2[384];
  r1[c] = s1; r2[c] = s1q;
  __syncthreads();
  if (c == 0) {
    float S = 0.f, Q = 0.f;
    for (int i = 0; i < 384; ++i) { S += r1[i]; Q += r2[i]; }
    float n = 4096.0f * 384.0f;
    float mu = S / n;
    stats[3072 + b] = mu;
    stats[3080 + b] = Q / n - mu * mu;
  }
}

__global__ __launch_bounds__(256) void adapter_out_k(const _Float16* __restrict__ xn,
                                                     const float* __restrict__ stats,
                                                     const float* __restrict__ cw,
                                                     const float* __restrict__ cb,
                                                     const float* __restrict__ sw,
                                                     const float* __restrict__ sb,
                                                     float* __restrict__ adpt) {
  long id = (long)blockIdx.x * 256 + threadIdx.x;
  if (id >= (long)NTOK * 384) return;
  int g = (int)(id / 384);
  int i = (int)(id % 384);
  int b = g >> 12;
  int m = g2m(g);
  float x0 = (float)xn[(size_t)m * DIMC + i];
  float x1 = (float)xn[(size_t)m * DIMC + 384 + i];
  float m0 = stats[b * 384 + i];
  float mu = stats[3072 + b];
  float vr = stats[3080 + b];
  float v0 = x0 * sigmoidf_(cw[i] * m0 + cb[i]);
  float gn = (x1 - mu) * rsqrtf(vr + 1e-5f);
  float v1 = x1 * sigmoidf_(sw[i] * gn + sb[i]);
  // channel_shuffle(groups=2): out[2i] = branch0[i], out[2i+1] = branch1[i]
  adpt[(size_t)g * DIMC + 2 * i]     = v0;
  adpt[(size_t)g * DIMC + 2 * i + 1] = v1;
}

// ---------------- generic WMMA GEMM with fused epilogues ----------------
// block = 8 waves; block tile 256x64; wave w owns rows [blk.x*256 + 32w, +32)
// (two 16-row A fragments share each B fragment -> 8 WMMAs per 32-K step).
// B tile (64x32 f16) staged through a 4-deep LDS ring via async global->LDS
// copies pipelined 2 tiles ahead (ASYNCcnt), one workgroup barrier per step.
// MODE 0: QKV       -> outh f16 [M][N], +bias
// MODE 1: adapter-1 -> outh f16, gelu(+bias)
// MODE 2: adapter-2 -> val = qkv_orig + aw*(acc+bias); scatter Q/K/V^T (outh base)
// MODE 3: proj      -> outf f32 = x + (acc+bias) + 0.5*adpt   (window->global map)
// MODE 4: mlp-1     -> outh f16, gelu(+bias)
// MODE 5: mlp-2     -> outf f32 += acc+bias
template <int MODE>
__global__ __launch_bounds__(256) void gemm_wmma_k(
    const _Float16* __restrict__ A, const _Float16* __restrict__ Wt,
    const float* __restrict__ bias, int M, int N, int K,
    const _Float16* __restrict__ auxh, const float* __restrict__ auxf,
    const float* __restrict__ auxf2, const float* __restrict__ awp,
    _Float16* __restrict__ outh, float* __restrict__ outf) {
  __shared__ __align__(16) _Float16 bsm[4][64][32];   // 4 x 4KB B tiles
  int t     = threadIdx.x;
  int lane  = t & 31;
  int wv    = t >> 5;
  int mbase = blockIdx.x * 256 + wv * 32;
  int nbase = blockIdx.y * 64;

  // async-copy assignment: thread t moves 16B: B row (= N col) t/4, chunk t%4
  int arow = t >> 2, achk = t & 3;
  const _Float16* gsrc = Wt + (size_t)(nbase + arow) * K + achk * 8;
  unsigned lds0 = ((unsigned)(size_t)(void*)&bsm[0][0][0]) +
                  (unsigned)(arow * 64 + achk * 16);

  int nsteps = K >> 5;
  // prologue: stage tiles 0 and 1
  async_g2l_b128(lds0, gsrc);
  if (nsteps > 1) async_g2l_b128(lds0 + 4096u, gsrc + 32);

  v8f acc[8] = {v8f{}, v8f{}, v8f{}, v8f{}, v8f{}, v8f{}, v8f{}, v8f{}};
  for (int s = 0; s < nsteps; ++s) {
    int k0 = s << 5;
    if (s + 2 < nsteps)
      async_g2l_b128(lds0 + (unsigned)(((s + 2) & 3) * 4096),
                     gsrc + k0 + 64);
    int remaining = nsteps - 1 - s;      // tiles issued after tile s
    if (remaining >= 2)      wait_async_le2();   // tile s has landed
    else if (remaining == 1) wait_async_le1();
    else                     wait_async_le0();
    __syncthreads();                     // all waves' async writes visible

    v16h a0 = load_frag_a(A, K, mbase,      k0, lane);
    v16h a1 = load_frag_a(A, K, mbase + 16, k0, lane);
    if (k0 + 64 < K)
      __builtin_prefetch(A + (size_t)(mbase + (lane & 15)) * K + k0 + 64, 0, 0);
    const _Float16* bb = &bsm[s & 3][0][0];
#pragma unroll
    for (int j = 0; j < 4; ++j) {
      v16h b = load_frag_b(bb, 32, j * 16, 0, lane);
      acc[j]     = wmma16x16x32(a0, b, acc[j]);
      acc[4 + j] = wmma16x16x32(a1, b, acc[4 + j]);
    }
    // no trailing barrier: next overwrite of this ring slot is 2 steps away,
    // separated from all readers by the next step's barrier
  }

  int col = lane & 15;
  int rb  = (lane >> 4) << 3;
#pragma unroll
  for (int mm = 0; mm < 2; ++mm) {
#pragma unroll
    for (int j = 0; j < 4; ++j) {
      int n = nbase + j * 16 + col;
      float bn = bias[n];
#pragma unroll
      for (int i = 0; i < 8; ++i) {
        int m = mbase + mm * 16 + rb + i;
        float v = acc[mm * 4 + j][i] + bn;
        if constexpr (MODE == 0) {
          outh[(size_t)m * N + n] = (_Float16)v;
        } else if constexpr (MODE == 1 || MODE == 4) {
          outh[(size_t)m * N + n] = (_Float16)gelu_erf(v);
        } else if constexpr (MODE == 2) {
          float q0  = (float)auxh[(size_t)m * N + n];
          float val = q0 + awp[0] * v;
          int win = m >> 8, pos = m & 255;
          int p = n / DIMC, rem = n % DIMC;
          int head = rem >> 6, d = rem & 63;
          size_t bh = (size_t)win * NHC + head;
          const size_t QSZ = (size_t)BHCNT * 256 * HDC;
          if (p == 0)
            outh[(bh * 256 + pos) * HDC + d] = (_Float16)val;              // Q
          else if (p == 1)
            outh[QSZ + (bh * 256 + pos) * HDC + d] = (_Float16)val;        // K
          else
            outh[2 * QSZ + (bh * HDC + d) * 256 + pos] = (_Float16)val;    // V^T
        } else if constexpr (MODE == 3) {
          int g = m2g(m);
          size_t idx = (size_t)g * DIMC + n;
          outf[idx] = auxf[idx] + v + 0.5f * auxf2[idx];
        } else if constexpr (MODE == 5) {
          size_t idx = (size_t)m * N + n;
          outf[idx] = outf[idx] + v;
        }
      }
    }
  }
}

// ---------------- window attention: one block per (window*head, 32-q chunk) --
__global__ __launch_bounds__(256) void attn_k(const _Float16* __restrict__ Qb,
                                              const _Float16* __restrict__ Kb,
                                              const _Float16* __restrict__ Vb,
                                              const float* __restrict__ relh,
                                              const float* __restrict__ relw,
                                              _Float16* __restrict__ Ob) {
  __shared__ float    S[32][256];
  __shared__ _Float16 P[32][256];
  __shared__ float    Ht[32][16];
  __shared__ float    Wt2[32][16];

  int bh = blockIdx.x;
  int q0 = blockIdx.y * 32;
  int head = bh % NHC;
  int win  = bh / NHC;
  int t = threadIdx.x, lane = t & 31, wv = t >> 5;

  const _Float16* Qrow = Qb + (size_t)bh * 256 * HDC;
  const _Float16* Krow = Kb + (size_t)bh * 256 * HDC;
  const _Float16* Vrow = Vb + (size_t)bh * HDC * 256;

  // decomposed relative-position bias terms
#pragma unroll
  for (int p = 0; p < 2; ++p) {
    int idx = t * 2 + p;           // 0..511 -> (mq 0..31, kk 0..15)
    int mq = idx >> 4, kk = idx & 15;
    int n = q0 + mq;
    int qh = n >> 4, qw = n & 15;
    const _Float16* qv = Qrow + (size_t)n * HDC;
    const float* rh = relh + (size_t)(qh - kk + 15) * HDC;
    const float* rw = relw + (size_t)(qw - kk + 15) * HDC;
    float sh = 0.f, sw = 0.f;
    for (int c = 0; c < HDC; ++c) {
      float qf = (float)qv[c];
      sh += qf * rh[c];
      sw += qf * rw[c];
    }
    Ht[mq][kk] = sh;
    Wt2[mq][kk] = sw;
  }
  __syncthreads();

  // S = scale * Q K^T + bias : 2 mt x 16 nt tiles, wave -> (mt = wv&1, 4 nt's)
  {
    int mt = wv & 1;
    int ntg = wv >> 1;
    v16h a0 = load_frag_a(Qrow, HDC, q0 + mt * 16, 0, lane);
    v16h a1 = load_frag_a(Qrow, HDC, q0 + mt * 16, 32, lane);
#pragma unroll
    for (int jj = 0; jj < 4; ++jj) {
      int nt = ntg * 4 + jj;
      v8f acc = {};
      v16h b0 = load_frag_b(Krow, HDC, nt * 16, 0, lane);
      v16h b1 = load_frag_b(Krow, HDC, nt * 16, 32, lane);
      acc = wmma16x16x32(a0, b0, acc);
      acc = wmma16x16x32(a1, b1, acc);
      int col = lane & 15;
      int rb  = (lane >> 4) << 3;
#pragma unroll
      for (int i = 0; i < 8; ++i) {
        int mr = mt * 16 + rb + i;
        // key position = nt*16+col -> kh = nt, kw = col
        S[mr][nt * 16 + col] = acc[i] * 0.125f + Ht[mr][nt] + Wt2[mr][col];
      }
    }
  }
  __syncthreads();

  // softmax: 8 lanes per row
  {
    int row = t >> 3, sub = t & 7;
    float mx = -1e30f;
    for (int c = sub * 32; c < sub * 32 + 32; ++c) mx = fmaxf(mx, S[row][c]);
    for (int msk = 1; msk < 8; msk <<= 1) mx = fmaxf(mx, __shfl_xor(mx, msk, 8));
    float sm = 0.f;
    for (int c = sub * 32; c < sub * 32 + 32; ++c) {
      float e = __expf(S[row][c] - mx);
      P[row][c] = (_Float16)e;
      sm += e;
    }
    for (int msk = 1; msk < 8; msk <<= 1) sm += __shfl_xor(sm, msk, 8);
    float inv = 1.0f / sm;
    for (int c = sub * 32; c < sub * 32 + 32; ++c)
      P[row][c] = (_Float16)((float)P[row][c] * inv);
  }
  __syncthreads();

  // O = P @ V : 2 mt x 4 dt tiles, one per wave
  {
    int mt = wv & 1;
    int dt = wv >> 1;
    v8f oacc = {};
    const _Float16* Pb = &P[0][0];
    for (int k0 = 0; k0 < 256; k0 += 32) {
      v16h pa = load_frag_a(Pb, 256, mt * 16, k0, lane);
      v16h vb = load_frag_b(Vrow, 256, dt * 16, k0, lane);
      oacc = wmma16x16x32(pa, vb, oacc);
    }
    int col = lane & 15;
    int rb  = (lane >> 4) << 3;
#pragma unroll
    for (int i = 0; i < 8; ++i) {
      int pos = q0 + mt * 16 + rb + i;
      int d = dt * 16 + col;
      Ob[((size_t)(win * 256 + pos) * NHC + head) * HDC + d] = (_Float16)oacc[i];
    }
  }
}

// ---------------- workspace layout (bytes) ----------------
static constexpr size_t OFF_WQ   = 0;
static constexpr size_t OFF_WA1  = OFF_WQ   + (size_t)DIMC * 3 * DIMC * 2;
static constexpr size_t OFF_WA2  = OFF_WA1  + (size_t)3 * DIMC * AD_C * 2;
static constexpr size_t OFF_WP   = OFF_WA2  + (size_t)AD_C * 3 * DIMC * 2;
static constexpr size_t OFF_WM1  = OFF_WP   + (size_t)DIMC * DIMC * 2;
static constexpr size_t OFF_WM2  = OFF_WM1  + (size_t)DIMC * MLPD_C * 2;
static constexpr size_t OFF_STAT = OFF_WM2  + (size_t)MLPD_C * DIMC * 2;
static constexpr size_t OFF_ADPT = OFF_STAT + (size_t)4096 * 4;
static constexpr size_t OFF_XN   = OFF_ADPT + (size_t)NTOK * DIMC * 4;   // also LN2
static constexpr size_t OFF_OB   = OFF_XN   + (size_t)NTOK * DIMC * 2;
static constexpr size_t OFF_QKV0 = OFF_OB   + (size_t)NTOK * DIMC * 2;   // also H1
static constexpr size_t OFF_HAD  = OFF_QKV0 + (size_t)NTOK * 3 * DIMC * 2;
static constexpr size_t OFF_QKVF = OFF_HAD  + (size_t)NTOK * AD_C * 2;

extern "C" void kernel_launch(void* const* d_in, const int* in_sizes, int n_in,
                              void* d_out, int out_size, void* d_ws, size_t ws_size,
                              hipStream_t stream) {
  (void)in_sizes; (void)n_in; (void)out_size; (void)ws_size;
  const float* x    = (const float*)d_in[0];
  const float* w1   = (const float*)d_in[1];
  const float* b1   = (const float*)d_in[2];
  const float* Wqkv = (const float*)d_in[3];
  const float* bqkv = (const float*)d_in[4];
  const float* A1   = (const float*)d_in[5];
  const float* ba1  = (const float*)d_in[6];
  const float* A2   = (const float*)d_in[7];
  const float* ba2  = (const float*)d_in[8];
  const float* awp  = (const float*)d_in[9];
  const float* relh = (const float*)d_in[10];
  const float* relw = (const float*)d_in[11];
  const float* Wp   = (const float*)d_in[12];
  const float* bp   = (const float*)d_in[13];
  const float* cw   = (const float*)d_in[14];
  const float* cb   = (const float*)d_in[15];
  const float* sw   = (const float*)d_in[16];
  const float* sb   = (const float*)d_in[17];
  const float* w2   = (const float*)d_in[18];
  const float* b2   = (const float*)d_in[19];
  const float* Wm1  = (const float*)d_in[20];
  const float* bm1  = (const float*)d_in[21];
  const float* Wm2  = (const float*)d_in[22];
  const float* bm2  = (const float*)d_in[23];
  float* out = (float*)d_out;
  char*  ws  = (char*)d_ws;

  _Float16* WQt  = (_Float16*)(ws + OFF_WQ);
  _Float16* WA1t = (_Float16*)(ws + OFF_WA1);
  _Float16* WA2t = (_Float16*)(ws + OFF_WA2);
  _Float16* WPt  = (_Float16*)(ws + OFF_WP);
  _Float16* WM1t = (_Float16*)(ws + OFF_WM1);
  _Float16* WM2t = (_Float16*)(ws + OFF_WM2);
  float*    STAT = (float*)(ws + OFF_STAT);
  float*    ADPT = (float*)(ws + OFF_ADPT);
  _Float16* XN   = (_Float16*)(ws + OFF_XN);    // LN1 out (win order) / LN2 out
  _Float16* OB   = (_Float16*)(ws + OFF_OB);
  _Float16* QKV0 = (_Float16*)(ws + OFF_QKV0);  // qkv_orig; later aliased by H1
  _Float16* HAD  = (_Float16*)(ws + OFF_HAD);
  _Float16* QKVF = (_Float16*)(ws + OFF_QKVF);  // Q | K | V^T
  _Float16* H1   = (_Float16*)(ws + OFF_QKV0);
  const size_t QSZ = (size_t)BHCNT * 256 * HDC;

  auto cvt = [&](const float* s, _Float16* d, int K, int N) {
    long tot = (long)K * N;
    cvt_wt_k<<<(int)((tot + 255) / 256), 256, 0, stream>>>(s, d, K, N);
  };
  cvt(Wqkv, WQt, DIMC, 3 * DIMC);
  cvt(A1, WA1t, 3 * DIMC, AD_C);
  cvt(A2, WA2t, AD_C, 3 * DIMC);
  cvt(Wp, WPt, DIMC, DIMC);
  cvt(Wm1, WM1t, DIMC, MLPD_C);
  cvt(Wm2, WM2t, MLPD_C, DIMC);

  // LN1 -> f16 (window-token order)
  ln_k<<<NTOK, 256, 0, stream>>>(x, w1, b1, XN, 1);

  // shuffle adapter
  adapter_stats_k<<<8, 384, 0, stream>>>(XN, STAT);
  adapter_out_k<<<(NTOK * 384) / 256, 256, 0, stream>>>(XN, STAT, cw, cb, sw, sb, ADPT);

  // qkv_orig = xn @ Wqkv + bqkv
  gemm_wmma_k<0><<<dim3(NTOK / 256, (3 * DIMC) / 64), 256, 0, stream>>>(
      XN, WQt, bqkv, NTOK, 3 * DIMC, DIMC,
      nullptr, nullptr, nullptr, nullptr, QKV0, nullptr);
  // h = gelu(qkv_orig @ A1 + ba1)
  gemm_wmma_k<1><<<dim3(NTOK / 256, AD_C / 64), 256, 0, stream>>>(
      QKV0, WA1t, ba1, NTOK, AD_C, 3 * DIMC,
      nullptr, nullptr, nullptr, nullptr, HAD, nullptr);
  // qkv = qkv_orig + aw*(h @ A2 + ba2), scattered into Q / K / V^T layouts
  gemm_wmma_k<2><<<dim3(NTOK / 256, (3 * DIMC) / 64), 256, 0, stream>>>(
      HAD, WA2t, ba2, NTOK, 3 * DIMC, AD_C,
      QKV0, nullptr, nullptr, awp, QKVF, nullptr);

  // windowed attention with decomposed rel-pos bias
  attn_k<<<dim3(BHCNT, 8), 256, 0, stream>>>(QKVF, QKVF + QSZ, QKVF + 2 * QSZ,
                                             relh, relw, OB);

  // y = x + (attn @ Wp + bp) + 0.5 * adapter_out   (window->global remap)
  gemm_wmma_k<3><<<dim3(NTOK / 256, DIMC / 64), 256, 0, stream>>>(
      OB, WPt, bp, NTOK, DIMC, DIMC,
      nullptr, x, ADPT, nullptr, nullptr, out);

  // LN2 -> f16 (global order, reuse XN buffer)
  ln_k<<<NTOK, 256, 0, stream>>>(out, w2, b2, XN, 0);

  // h1 = gelu(ln2 @ Wm1 + bm1)
  gemm_wmma_k<4><<<dim3(NTOK / 256, MLPD_C / 64), 256, 0, stream>>>(
      XN, WM1t, bm1, NTOK, MLPD_C, DIMC,
      nullptr, nullptr, nullptr, nullptr, H1, nullptr);
  // out += h1 @ Wm2 + bm2
  gemm_wmma_k<5><<<dim3(NTOK / 256, DIMC / 64), 256, 0, stream>>>(
      H1, WM2t, bm2, NTOK, DIMC, MLPD_C,
      nullptr, nullptr, nullptr, nullptr, nullptr, out);
}
